// MC_GRU_37211596653410
// MI455X (gfx1250) — matrix operand order repaired and checked by reading, not compiled, not run on hardware
//
#include <hip/hip_runtime.h>
#include <hip/hip_bf16.h>

typedef __attribute__((ext_vector_type(2)))  float    v2f;
typedef __attribute__((ext_vector_type(8)))  float    v8f;
typedef __attribute__((ext_vector_type(16))) _Float16 v16h;

#define F_FEAT 16
#define D_IN   4
#define HID    32
#define BATCH  1024
#define TMAX   512
#define G3H    96   // 3*HID gate rows

#define L2E 1.442695041f   // log2(e)

// Raw-hardware sigmoid pieces: bare v_exp_f32 / v_rcp_f32, no OCML range guards
// or Newton refinement (gate arguments are well-bounded).
__device__ __forceinline__ float fast_sig_arg(float arg /* = -L2E*x (+pre-bias) */) {
    return __builtin_amdgcn_rcpf(1.0f + __builtin_amdgcn_exp2f(arg));
}

// Pure-VALU cross-lane xor within wave32, no LDS:
//   XOR 16 -> v_permlanex16_b32 (identity selectors)
//   XOR 1/2/4/8 -> v_permlane16_b32 with per-offset selector tables
__device__ __forceinline__ float xhalf_swap(float x) {
    unsigned u = __float_as_uint(x);
    unsigned r = __builtin_amdgcn_permlanex16(u, u, 0x76543210u, 0xFEDCBA98u, false, false);
    return __uint_as_float(r);
}
template <unsigned SELLO, unsigned SELHI>
__device__ __forceinline__ float lane16_perm(float x) {
    unsigned u = __float_as_uint(x);
    unsigned r = __builtin_amdgcn_permlane16(u, u, SELLO, SELHI, false, false);
    return __uint_as_float(r);
}
__device__ __forceinline__ float xor1(float x) { return lane16_perm<0x67452301u, 0xEFCDAB89u>(x); }
__device__ __forceinline__ float xor2(float x) { return lane16_perm<0x54761032u, 0xDCFE98BAu>(x); }
__device__ __forceinline__ float xor4(float x) { return lane16_perm<0x32107654u, 0xBA98FEDCu>(x); }
__device__ __forceinline__ float xor8(float x) { return lane16_perm<0xFEDCBA98u, 0x76543210u>(x); }

// One wave32 per (feature f, 16-batch tile). Everything is computed transposed:
//   gi^T = W_ih[96x4]  * x^T[4x16]   via V_WMMA_F32_16X16X4_F32   (6 gate-row tiles)
//   gh^T = W_hh[96x32] * h^T[32x16]  via V_WMMA_F32_16X16X32_F16  (6 gate-row tiles)
// C/D layout: lane = batch column, VGPR index = gate/hidden row (mod 8, +8 for hi half).
// All WMMA chains start from inline C=0 (no SRC2 register read -> no bias-copy movs);
// biases are folded into the gate-math FMAs, pre-scaled by -log2(e) where possible.
__global__ __launch_bounds__(128) void mc_gru_recurrent(
    const float* __restrict__ input,   // [B, F, T, D]
    const int*   __restrict__ lens,    // [B]
    const float* __restrict__ W_ih,    // [F, 96, 4]
    const float* __restrict__ W_hh,    // [F, 96, 32]
    const float* __restrict__ b_ih,    // [F, 96]
    const float* __restrict__ b_hh,    // [F, 96]
    float*       __restrict__ hcat)    // [B, F*H] feature-major
{
    const int lane = threadIdx.x & 31;
    const int wid  = (int)((blockIdx.x * blockDim.x + threadIdx.x) >> 5);
    const int f    = wid >> 6;                  // 0..15
    const int b0   = (wid & 63) << 4;           // batch tile base (16 rows)
    const int row  = lane & 15;                 // M row / batch column within tile
    const int hi16 = lane >> 4;                 // lane half

    // ---- Constant A operands, resident in registers for the whole sequence ----
    v2f  wihA[6];   // W_ih tiles, 16x4 f32 A layout: lane=row, vgpr0:K=0|2, vgpr1:K=1|3
    v16h whhA[6];   // W_hh tiles, 16x32 f16 A layout
    #pragma unroll
    for (int j = 0; j < 6; ++j) {
        const float* wi = W_ih + ((size_t)f * G3H + 16 * j + row) * D_IN + 2 * hi16;
        wihA[j][0] = wi[0];
        wihA[j][1] = wi[1];
        const float* wh = W_hh + ((size_t)f * G3H + 16 * j + row) * HID;
        #pragma unroll
        for (int e = 0; e < 16; ++e) {
            // f16 A layout: element e -> K = e (+16 if hi half)
            int k = e + (hi16 ? 16 : 0);
            whhA[j][e] = (_Float16)wh[k];
        }
    }

    // ---- Biases in C layout (row = 16j + v + 8*hi16), pre-scaled for the gate FMAs ----
    v8f brz[4];     // r,z tiles: -log2e * (b_ih + b_hh)   -> exp2 argument offset
    v8f bin[2];     // n tiles:  b_ih (raw)
    v8f bhn[2];     // n tiles:  b_hh (raw)
    #pragma unroll
    for (int j = 0; j < 4; ++j)
        #pragma unroll
        for (int v = 0; v < 8; ++v) {
            int m = 16 * j + v + 8 * hi16;
            brz[j][v] = -L2E * (b_ih[f * G3H + m] + b_hh[f * G3H + m]);
        }
    #pragma unroll
    for (int j = 0; j < 2; ++j)
        #pragma unroll
        for (int v = 0; v < 8; ++v) {
            int m = 16 * (j + 4) + v + 8 * hi16;
            bin[j][v] = b_ih[f * G3H + m];
            bhn[j][v] = b_hh[f * G3H + m];
        }

    const int len_lane = lens[b0 + row];
    // Wave-max sequence length over the 16 batch rows: every lane loads all 16
    // lens (4x int4, 64B-aligned) and maxes locally -> wave-uniform, no LDS.
    int tmax_w = 1;
    {
        const int4* lp = (const int4*)(lens + b0);
        #pragma unroll
        for (int q = 0; q < 4; ++q) {
            int4 l = lp[q];
            tmax_w = max(tmax_w, max(max(l.x, l.y), max(l.z, l.w)));
        }
    }

    // Hidden state h^T[32 x 16] as two 16x16 f32 C-layout tiles (rows 0-15 / 16-31)
    v8f ht0 = {};
    v8f ht1 = {};
    const v8f zeroC = {};

    const float* xbase = input + (((size_t)(b0 + row) * F_FEAT + f) * TMAX) * D_IN + 2 * hi16;

    // Software-pipelined x stream (input is read exactly once -> nontemporal).
    v2f xB = __builtin_nontemporal_load((const v2f*)xbase);

    for (int t = 0; t < tmax_w; ++t) {
        // prefetch next timestep's x (clamped; stays in-bounds)
        const int tn = (t + 1 < tmax_w) ? (t + 1) : t;
        v2f xNext = __builtin_nontemporal_load((const v2f*)(xbase + (size_t)tn * D_IN));

        // ---- rebuild h^T as f16 B operand (32x16): pure-VALU half-wave swap
        v16h hB;
        #pragma unroll
        for (int v = 0; v < 8; ++v) {
            float a0 = ht0[v], a1 = ht1[v];
            float s0 = xhalf_swap(a0);
            float s1 = xhalf_swap(a1);
            // element e<8:  K = e    (lo half: rows 0-7 own ht0)   / K=16+e (hi half: partner ht1)
            // element e>=8: K = 8+e' (lo half: partner ht0)        / K=24+e' (hi half: own ht1)
            hB[v]     = (_Float16)(hi16 ? s1 : a0);
            hB[v + 8] = (_Float16)(hi16 ? a1 : s0);
        }

        // ---- gh^T: 6x V_WMMA_F32_16X16X32_F16, inline C=0 (no SRC2 copies)
        v8f aR0 = __builtin_amdgcn_wmma_f32_16x16x32_f16(false, whhA[0], false, hB, (short)0, zeroC, false, false);
        v8f aR1 = __builtin_amdgcn_wmma_f32_16x16x32_f16(false, whhA[1], false, hB, (short)0, zeroC, false, false);
        v8f aZ0 = __builtin_amdgcn_wmma_f32_16x16x32_f16(false, whhA[2], false, hB, (short)0, zeroC, false, false);
        v8f aZ1 = __builtin_amdgcn_wmma_f32_16x16x32_f16(false, whhA[3], false, hB, (short)0, zeroC, false, false);
        v8f hN0 = __builtin_amdgcn_wmma_f32_16x16x32_f16(false, whhA[4], false, hB, (short)0, zeroC, false, false);
        v8f hN1 = __builtin_amdgcn_wmma_f32_16x16x32_f16(false, whhA[5], false, hB, (short)0, zeroC, false, false);

        // ---- gi^T accumulated on top: 6x V_WMMA_F32_16X16X4_F32 (in-place chains)
        aR0 = __builtin_amdgcn_wmma_f32_16x16x4_f32(false, wihA[0], false, xB, (short)0, aR0, false, false);
        aR1 = __builtin_amdgcn_wmma_f32_16x16x4_f32(false, wihA[1], false, xB, (short)0, aR1, false, false);
        aZ0 = __builtin_amdgcn_wmma_f32_16x16x4_f32(false, wihA[2], false, xB, (short)0, aZ0, false, false);
        aZ1 = __builtin_amdgcn_wmma_f32_16x16x4_f32(false, wihA[3], false, xB, (short)0, aZ1, false, false);
        v8f aN0 = __builtin_amdgcn_wmma_f32_16x16x4_f32(false, wihA[4], false, xB, (short)0, zeroC, false, false);
        v8f aN1 = __builtin_amdgcn_wmma_f32_16x16x4_f32(false, wihA[5], false, xB, (short)0, zeroC, false, false);

        // ---- gates + packed-sequence mask (lane's batch column frozen past its length)
        const bool live = (t < len_lane);
        #pragma unroll
        for (int v = 0; v < 8; ++v) {
            // sigmoid(x + b) = rcp(1 + exp2(-L2E*x + pre_b)), pre_b = -L2E*b
            float r0 = fast_sig_arg(fmaf(-L2E, aR0[v], brz[0][v]));
            float r1 = fast_sig_arg(fmaf(-L2E, aR1[v], brz[1][v]));
            float z0 = fast_sig_arg(fmaf(-L2E, aZ0[v], brz[2][v]));
            float z1 = fast_sig_arg(fmaf(-L2E, aZ1[v], brz[3][v]));
            // n = tanh(i_n + b_in + r*(h_n + b_hn)) = 2*sigmoid(2y) - 1
            float y0 = fmaf(r0, hN0[v] + bhn[0][v], aN0[v] + bin[0][v]);
            float y1 = fmaf(r1, hN1[v] + bhn[1][v], aN1[v] + bin[1][v]);
            float n0 = fmaf(2.0f, fast_sig_arg(-2.0f * L2E * y0), -1.0f);
            float n1 = fmaf(2.0f, fast_sig_arg(-2.0f * L2E * y1), -1.0f);
            // h' = n + z*(h - n)
            float h0 = fmaf(z0, ht0[v] - n0, n0);
            float h1 = fmaf(z1, ht1[v] - n1, n1);
            ht0[v] = live ? h0 : ht0[v];
            ht1[v] = live ? h1 : ht1[v];
        }

        xB = xNext;
    }

    // ---- write h_cat[b, f*H + hidden] (feature-major concat, matches reference)
    float* out = hcat + (size_t)(b0 + row) * (F_FEAT * HID) + f * HID;
    #pragma unroll
    for (int v = 0; v < 8; ++v) {
        out[v + 8 * hi16]      = ht0[v];
        out[16 + v + 8 * hi16] = ht1[v];
    }
}

// Head: relu(h_cat @ W_sq^T + b_sq) -> sigmoid(z1 @ W_out^T + b_out).
// One wave per batch row: lane j owns hidden unit j's 512-wide dot, then a
// pure-VALU permlane xor tree reduces the 32-wide output dot.
__global__ __launch_bounds__(256) void mc_gru_head(
    const float* __restrict__ hcat,   // [B, 512]
    const float* __restrict__ W_sq,   // [32, 512]
    const float* __restrict__ b_sq,   // [32]
    const float* __restrict__ W_out,  // [1, 32]
    const float* __restrict__ b_out,  // [1]
    float*       __restrict__ out)    // [B]
{
    const int lane = threadIdx.x & 31;
    const int b    = (int)((blockIdx.x * blockDim.x + threadIdx.x) >> 5);

    const float* hrow = hcat + (size_t)b * (F_FEAT * HID);
    const float* wrow = W_sq + (size_t)lane * (F_FEAT * HID);
    float acc = 0.0f;
    #pragma unroll 4
    for (int k = 0; k < F_FEAT * HID; k += 4) {
        acc = fmaf(wrow[k + 0], hrow[k + 0], acc);
        acc = fmaf(wrow[k + 1], hrow[k + 1], acc);
        acc = fmaf(wrow[k + 2], hrow[k + 2], acc);
        acc = fmaf(wrow[k + 3], hrow[k + 3], acc);
    }
    float z1 = fmaxf(acc + b_sq[lane], 0.0f);
    float p  = z1 * W_out[lane];
    p += xor1(p);
    p += xor2(p);
    p += xor4(p);
    p += xor8(p);
    p += xhalf_swap(p);
    if (lane == 0)
        out[b] = fast_sig_arg(-L2E * (p + b_out[0]));
}

extern "C" void kernel_launch(void* const* d_in, const int* in_sizes, int n_in,
                              void* d_out, int out_size, void* d_ws, size_t ws_size,
                              hipStream_t stream) {
    const float* input = (const float*)d_in[0];
    const int*   lens  = (const int*)  d_in[1];
    const float* W_ih  = (const float*)d_in[2];
    const float* W_hh  = (const float*)d_in[3];
    const float* b_ih  = (const float*)d_in[4];
    const float* b_hh  = (const float*)d_in[5];
    const float* W_sq  = (const float*)d_in[6];
    const float* b_sq  = (const float*)d_in[7];
    const float* W_out = (const float*)d_in[8];
    const float* b_out = (const float*)d_in[9];

    float* hcat = (float*)d_ws;   // B * F*H * 4 = 2 MB scratch

    // 1024 waves total: F(16) x B/16(64). 128 threads = 4 waves/block -> 256 blocks.
    mc_gru_recurrent<<<256, 128, 0, stream>>>(input, lens, W_ih, W_hh, b_ih, b_hh, hcat);
    // 1024 waves: one per batch row. 256 threads = 8 waves/block -> 128 blocks.
    mc_gru_head<<<128, 256, 0, stream>>>(hcat, W_sq, b_sq, W_out, b_out, (float*)d_out);
}
